// ChunkBoundaryPredictor_27221502722146
// MI455X (gfx1250) — matrix operand order repaired and checked by reading, not compiled, not run on hardware
//
#include <hip/hip_runtime.h>
#include <hip/hip_bf16.h>
#include <cstddef>

#define T_LEN 65536
#define HID   128
#define G4    512          // 4*HID gate width
#define WROW  132          // LDS row stride for w_hh (16B aligned, bank-spread)

typedef __attribute__((ext_vector_type(16))) _Float16 v16h;
typedef __attribute__((ext_vector_type(8)))  float    v8f;

__device__ __forceinline__ float sigf(float x) {
    return 1.0f / (1.0f + __expf(-x));
}

// ---------------------------------------------------------------------------
// Kernel 1: xg[t][n] = sum_k emb[tok[t]][k] * w_ih[n][k] + (b_ih[n]+b_hh[n])
// WMMA f32 = f16 x f16 tiles. Block = 256 threads = 8 waves; each wave owns a
// 16-wide N tile; block covers M-tile of 16 tokens x N range of 128.
// ---------------------------------------------------------------------------
__global__ __launch_bounds__(256) void xg_gemm_kernel(
    const int*   __restrict__ tokens,
    const float* __restrict__ emb,     // VOCAB x 128
    const float* __restrict__ w_ih,    // 512 x 128
    const float* __restrict__ b_ih,    // 512
    const float* __restrict__ b_hh,    // 512
    float*       __restrict__ xg)      // T x 512
{
    __shared__ _Float16 Xs[16][HID];   // 4 KB
    __shared__ _Float16 Ws[128][HID];  // 32 KB

    const int t0    = blockIdx.x * 16;
    const int nbase = blockIdx.y * 128;
    const int tid   = threadIdx.x;

    // Stage X tile (16 tokens x 128) as f16
    for (int i = tid; i < 16 * HID / 4; i += 256) {
        const int row = i / (HID / 4);
        const int col = (i % (HID / 4)) * 4;
        const int tok = tokens[t0 + row];
        const float4 v = *(const float4*)(emb + (size_t)tok * HID + col);
        Xs[row][col + 0] = (_Float16)v.x;
        Xs[row][col + 1] = (_Float16)v.y;
        Xs[row][col + 2] = (_Float16)v.z;
        Xs[row][col + 3] = (_Float16)v.w;
    }
    // Stage W tile (128 gate rows x 128) as f16
    for (int i = tid; i < 128 * HID / 4; i += 256) {
        const int row = i / (HID / 4);
        const int col = (i % (HID / 4)) * 4;
        const float4 v = *(const float4*)(w_ih + (size_t)(nbase + row) * HID + col);
        Ws[row][col + 0] = (_Float16)v.x;
        Ws[row][col + 1] = (_Float16)v.y;
        Ws[row][col + 2] = (_Float16)v.z;
        Ws[row][col + 3] = (_Float16)v.w;
    }
    __syncthreads();

    const int wave = tid >> 5;          // wave32
    const int lane = tid & 31;
    const int nrel = wave * 16;         // wave's N sub-tile within block
    const int r    = lane & 15;
    const bool hi  = lane >= 16;

    v8f c = {};
    #pragma unroll
    for (int kb = 0; kb < 4; ++kb) {
        const int k0 = kb * 32;
        v16h a, b;
        // A 16x32 f16 layout (ISA 7.12.2): lanes<16 rows hold K 0..7 & 16..23,
        // lanes>=16 hold K 8..15 & 24..31.
        const int ka = k0 + (hi ? 8 : 0);
        #pragma unroll
        for (int i = 0; i < 8; ++i) {
            a[i]     = Xs[r][ka + i];
            a[8 + i] = Xs[r][ka + 16 + i];
        }
        // B 32x16 f16: lane = column; lanes<16 hold K 0..7 & 16..23, lanes>=16 K 8..15 & 24..31.
        const int kbb = k0 + (hi ? 8 : 0);
        #pragma unroll
        for (int i = 0; i < 8; ++i) {
            b[i]     = Ws[nrel + r][kbb + i];
            b[8 + i] = Ws[nrel + r][kbb + 16 + i];
        }

        c = __builtin_amdgcn_wmma_f32_16x16x32_f16(
                /*neg_a=*/false, a, /*neg_b=*/false, b,
                /*c_mod=*/(short)0, c, /*reuse_a=*/false, /*reuse_b=*/false);
    }

    const int n     = nbase + nrel + r;
    const float bia = b_ih[n] + b_hh[n];
    const int mofs  = hi ? 8 : 0;      // C: VGPR v -> row v (lanes<16) / v+8
    #pragma unroll
    for (int vr = 0; vr < 8; ++vr) {
        xg[(size_t)(t0 + vr + mofs) * G4 + n] = c[vr] + bia;
    }
}

// ---------------------------------------------------------------------------
// Kernel 2: sequential LSTM scan, one persistent workgroup per direction.
// w_hh lives in LDS (512 x 132-padded f32 = 264 KB of the 320 KB WGP LDS).
// 512 threads: each computes one gate; threads 0..127 apply the cell update.
// ---------------------------------------------------------------------------
__global__ __launch_bounds__(512) void lstm_scan_kernel(
    const float* __restrict__ xg_f,   // T x 512 (with biases folded in)
    const float* __restrict__ xg_r,
    const float* __restrict__ w_hh_f, // 512 x 128
    const float* __restrict__ w_hh_r,
    float*       __restrict__ h_f,    // T x 128
    float*       __restrict__ h_r)
{
    extern __shared__ float smem[];
    float* Wl = smem;                    // 512 * WROW
    float* hl = smem + (size_t)G4 * WROW; // 128
    float* gl = hl + HID;                // 512

    const int dir = blockIdx.x;          // 0 = forward, 1 = reverse
    const float* __restrict__ xg   = dir ? xg_r  : xg_f;
    const float* __restrict__ w_hh = dir ? w_hh_r : w_hh_f;
    float*       __restrict__ hout = dir ? h_r   : h_f;

    const int j = threadIdx.x;

    // Load this thread's recurrent weight row into LDS (padded stride).
    {
        const float4* src = (const float4*)(w_hh + (size_t)j * HID);
        float*        dst = Wl + (size_t)j * WROW;
        #pragma unroll
        for (int k = 0; k < HID / 4; ++k) {
            const float4 v = src[k];
            dst[4 * k + 0] = v.x;
            dst[4 * k + 1] = v.y;
            dst[4 * k + 2] = v.z;
            dst[4 * k + 3] = v.w;
        }
    }
    if (j < HID) hl[j] = 0.0f;
    float cstate = 0.0f;
    __syncthreads();

    const float* __restrict__ wr = Wl + (size_t)j * WROW;

    for (int t = 0; t < T_LEN; ++t) {
        const int tt = dir ? (T_LEN - 1 - t) : t;

        float acc = xg[(size_t)tt * G4 + j];
        #pragma unroll 16
        for (int k = 0; k < HID; ++k)
            acc += wr[k] * hl[k];
        gl[j] = acc;
        __syncthreads();

        if (j < HID) {
            const float ig = sigf(gl[j]);
            const float fg = sigf(gl[HID + j]);
            const float gg = tanhf(gl[2 * HID + j]);
            const float og = sigf(gl[3 * HID + j]);
            cstate = fg * cstate + ig * gg;
            const float hj = og * tanhf(cstate);
            hl[j] = hj;
            hout[(size_t)tt * HID + j] = hj;
        }
        __syncthreads();
    }
}

// ---------------------------------------------------------------------------
// Kernel 3: scores = sigmoid(h_f . w_out[:128] + h_r . w_out[128:] + b_out)
// ---------------------------------------------------------------------------
__global__ __launch_bounds__(256) void head_kernel(
    const float* __restrict__ h_f,
    const float* __restrict__ h_r,
    const float* __restrict__ w_out,  // 1 x 256
    const float* __restrict__ b_out,  // 1
    float*       __restrict__ out)    // T
{
    const int t = blockIdx.x * blockDim.x + threadIdx.x;
    if (t >= T_LEN) return;
    const float* a = h_f + (size_t)t * HID;
    const float* b = h_r + (size_t)t * HID;
    float acc = b_out[0];
    #pragma unroll 8
    for (int k = 0; k < HID; ++k)
        acc += a[k] * w_out[k] + b[k] * w_out[HID + k];
    out[t] = sigf(acc);
}

// ---------------------------------------------------------------------------
extern "C" void kernel_launch(void* const* d_in, const int* in_sizes, int n_in,
                              void* d_out, int out_size, void* d_ws, size_t ws_size,
                              hipStream_t stream) {
    (void)in_sizes; (void)n_in; (void)out_size; (void)ws_size;

    const int*   tokens  = (const int*)  d_in[0];
    const float* emb     = (const float*)d_in[1];
    const float* w_ih_f  = (const float*)d_in[2];
    const float* w_hh_f  = (const float*)d_in[3];
    const float* b_ih_f  = (const float*)d_in[4];
    const float* b_hh_f  = (const float*)d_in[5];
    const float* w_ih_r  = (const float*)d_in[6];
    const float* w_hh_r  = (const float*)d_in[7];
    const float* b_ih_r  = (const float*)d_in[8];
    const float* b_hh_r  = (const float*)d_in[9];
    const float* w_out   = (const float*)d_in[10];
    const float* b_out   = (const float*)d_in[11];

    float* ws   = (float*)d_ws;
    float* xg_f = ws;                                   // T*512
    float* xg_r = xg_f + (size_t)T_LEN * G4;            // T*512
    float* h_f  = xg_r + (size_t)T_LEN * G4;            // T*128
    float* h_r  = h_f  + (size_t)T_LEN * HID;           // T*128

    // 1) Input projections via WMMA
    dim3 ggrid(T_LEN / 16, G4 / 128, 1);
    xg_gemm_kernel<<<ggrid, 256, 0, stream>>>(tokens, emb, w_ih_f, b_ih_f, b_hh_f, xg_f);
    xg_gemm_kernel<<<ggrid, 256, 0, stream>>>(tokens, emb, w_ih_r, b_ih_r, b_hh_r, xg_r);

    // 2) Both directional scans concurrently (one WGP each), w_hh in 264KB LDS
    const size_t lds_bytes = ((size_t)G4 * WROW + HID + G4) * sizeof(float);
    lstm_scan_kernel<<<2, 512, lds_bytes, stream>>>(xg_f, xg_r, w_hh_f, w_hh_r, h_f, h_r);

    // 3) Output head
    head_kernel<<<T_LEN / 256, 256, 0, stream>>>(h_f, h_r, w_out, b_out, (float*)d_out);
}
